// ArchitectureController_69432441307502
// MI455X (gfx1250) — matrix operand order repaired and checked by reading, not compiled, not run on hardware
//
#include <hip/hip_runtime.h>
#include <hip/hip_bf16.h>

// ---------------------------------------------------------------------------
// Architecture-controller LSTM sampler for MI455X (gfx1250, wave32, WMMA).
//   - bf16 WMMA 16x16x32 with f32 accumulate for all LSTM gate GEMMs
//   - weights pre-swizzled to B-fragment layout (L2-resident, coalesced b128)
//   - h (bf16) / c (f32) state LDS-resident per 16-row wave tile
// ---------------------------------------------------------------------------

typedef __bf16 bf16_t;
typedef __attribute__((ext_vector_type(16))) __bf16 v16bf;
typedef __attribute__((ext_vector_type(8)))  float  v8f;

#define HDIM   128
#define BATCH  65536
#define NDSTEP 5
#define NCSTEP 8

union FragBF {
    v16bf v;
    int4  q[2];
};

__device__ __forceinline__ float sigm(float x) {
    return 1.0f / (1.0f + __expf(-x));
}
__device__ __forceinline__ float tanh_fast(float x) {
    float ax = fabsf(x);
    float e  = __expf(-2.0f * ax);
    float t  = (1.0f - e) / (1.0f + e);
    return (x < 0.0f) ? -t : t;
}

// ---------------------------------------------------------------------------
// Prep: fp32 -> bf16 weight conversion into WMMA B-fragment swizzled layout.
// Element id e = ((((l*2+s)*32 + nt)*4 + kt)*32 + lane)*16 + j   (18 bits)
//   j = r*2+sub ; n = nt*16 + lane%16 ; k = kt*32 + (lane/16)*16 + 2r + sub
// Main kernel then reads lane-contiguous 32B per fragment.
// ---------------------------------------------------------------------------
__global__ __launch_bounds__(256)
void ctrl_prep_kernel(const float* __restrict__ Wih, const float* __restrict__ Whh,
                      const float* __restrict__ bih, const float* __restrict__ bhh,
                      bf16_t* __restrict__ wsw, float* __restrict__ wsb)
{
    int e = blockIdx.x * 256 + threadIdx.x;            // 0 .. 262143
    int j    =  e        & 15;
    int lane = (e >> 4)  & 31;
    int kt   = (e >> 9)  & 3;
    int nt   = (e >> 11) & 31;
    int s    = (e >> 16) & 1;
    int l    = (e >> 17) & 1;
    int r = j >> 1, sub = j & 1;
    int n = nt * 16 + (lane & 15);
    int k = kt * 32 + (lane >> 4) * 16 + 2 * r + sub;
    const float* W = s ? Whh : Wih;
    wsw[e] = (bf16_t)W[(l * 512 + n) * HDIM + k];
    if (e < 1024) wsb[e] = bih[e] + bhh[e];            // combined bias [l][4H]
}

// ---------------------------------------------------------------------------
// Main kernel: one wave (32 threads) per block, 16 batch rows per wave.
// ---------------------------------------------------------------------------
__global__ __launch_bounds__(32)
void ctrl_main_kernel(const float* __restrict__ emb,     // [5][5][128]
                      const float* __restrict__ dh_w,    // [5][5][128]
                      const float* __restrict__ dh_b,    // [5][5]
                      const float* __restrict__ ch_w,    // [8][2][128]
                      const float* __restrict__ ch_b,    // [8][2]
                      const float* __restrict__ minv,    // [8]
                      const float* __restrict__ maxv,    // [8]
                      const float* __restrict__ gumbel,  // [5][B][5]
                      const float* __restrict__ eps,     // [8][B]
                      const bf16_t* __restrict__ wsw,    // swizzled weights
                      const float* __restrict__ wsb,     // combined bias [2][512]
                      float* __restrict__ out)           // [14*B]
{
    __shared__ __align__(16) bf16_t xbuf[16 * HDIM];          // input x (bf16, A layout = row major)
    __shared__ __align__(16) bf16_t hbuf[2][2][16 * HDIM];    // [layer][parity]
    __shared__ __align__(16) float  cbuf[2][16 * HDIM];       // cell state f32
    __shared__ int chs[16];

    const int lane  = threadIdx.x;
    const int half  = lane >> 4;     // K-half selector for A/B fragments
    const int ncol  = lane & 15;     // N column of B/C fragments; also row id for A/head
    const int rowg0 = blockIdx.x * 16;

    for (int i = lane; i < 16 * HDIM; i += 32) {
        xbuf[i] = (bf16_t)0.0f;
        hbuf[0][0][i] = (bf16_t)0.0f; hbuf[0][1][i] = (bf16_t)0.0f;
        hbuf[1][0][i] = (bf16_t)0.0f; hbuf[1][1][i] = (bf16_t)0.0f;
        cbuf[0][i] = 0.0f; cbuf[1][i] = 0.0f;
    }
    __syncthreads();

    int pp0 = 0, pp1 = 0;
    float lp_acc = 0.0f;

    for (int t = 0; t < NDSTEP + NCSTEP; ++t) {
        // ---------------- LSTM: 2 layers -----------------------------------
        #pragma unroll
        for (int l = 0; l < 2; ++l) {
            const bf16_t* inp  = (l == 0) ? xbuf : hbuf[0][pp0];   // pp0 flipped after layer0
            const bf16_t* hin  = (l == 0) ? hbuf[0][pp0] : hbuf[1][pp1];
            bf16_t*       hout = (l == 0) ? hbuf[0][pp0 ^ 1] : hbuf[1][pp1 ^ 1];

            for (int jt = 0; jt < 8; ++jt) {           // 8 hidden tiles of 16
                v8f acc[4];
                #pragma unroll
                for (int g = 0; g < 4; ++g) {
                    float b = wsb[l * 512 + g * 128 + jt * 16 + ncol];
                    #pragma unroll
                    for (int r = 0; r < 8; ++r) acc[g][r] = b;
                }
                const bf16_t* rowx = inp + ncol * HDIM;
                const bf16_t* rowh = hin + ncol * HDIM;
                #pragma unroll
                for (int kt = 0; kt < 4; ++kt) {       // K = 128 in 4 tiles of 32
                    FragBF ax, ah;
                    int o0 = kt * 32 + half * 16;      // regs 0-3: k = half*16 + [0,8)
                    int o1 = o0 + 8;                   // regs 4-7: k = half*16 + 8 + [0,8)
                    ax.q[0] = *(const int4*)(rowx + o0);
                    ax.q[1] = *(const int4*)(rowx + o1);
                    ah.q[0] = *(const int4*)(rowh + o0);
                    ah.q[1] = *(const int4*)(rowh + o1);
                    #pragma unroll
                    for (int g = 0; g < 4; ++g) {
                        int nt = g * 8 + jt;           // n-tile of gate column block
                        const bf16_t* b0 = wsw + ((((l * 2 + 0) * 32 + nt) * 4 + kt) << 9) + lane * 16;
                        const bf16_t* b1 = wsw + ((((l * 2 + 1) * 32 + nt) * 4 + kt) << 9) + lane * 16;
                        FragBF bih, bhh;
                        bih.q[0] = *(const int4*)(b0);
                        bih.q[1] = *(const int4*)(b0 + 8);
                        bhh.q[0] = *(const int4*)(b1);
                        bhh.q[1] = *(const int4*)(b1 + 8);
                        acc[g] = __builtin_amdgcn_wmma_f32_16x16x32_bf16(
                            false, ax.v, false, bih.v, (short)0, acc[g], false, false);
                        acc[g] = __builtin_amdgcn_wmma_f32_16x16x32_bf16(
                            false, ah.v, false, bhh.v, (short)0, acc[g], false, false);
                    }
                }
                // elementwise gate update; C layout: M = r + half*8, N = ncol
                int j = jt * 16 + ncol;
                #pragma unroll
                for (int r = 0; r < 8; ++r) {
                    int m = r + half * 8;
                    float iv = sigm(acc[0][r]);
                    float fv = sigm(acc[1][r]);
                    float gv = tanh_fast(acc[2][r]);
                    float ov = sigm(acc[3][r]);
                    float cold = cbuf[l][m * HDIM + j];
                    float cn = fv * cold + iv * gv;
                    cbuf[l][m * HDIM + j] = cn;
                    hout[m * HDIM + j] = (bf16_t)(ov * tanh_fast(cn));
                }
            }
            __syncthreads();
            if (l == 0) pp0 ^= 1; else pp1 ^= 1;
        }

        const bf16_t* outrow = hbuf[1][pp1] + ncol * HDIM;   // new h1, row = ncol

        if (t < NDSTEP) {
            // -------- decision head: masked gumbel argmax ------------------
            if (lane < 16) {
                int rowg = rowg0 + ncol;
                int cnt = (t == 1) ? 4 : ((t == 4) ? 5 : 3);   // COUNTS
                float logits[5];
                for (int ci = 0; ci < 5; ++ci) {
                    float s = dh_b[t * 5 + ci];
                    if (ci < cnt) {
                        const float* w = dh_w + (t * 5 + ci) * HDIM;
                        float a = 0.0f;
                        for (int k = 0; k < HDIM; ++k) a += (float)outrow[k] * w[k];
                        s += a;
                    }
                    logits[ci] = s;
                }
                float mx = -1e30f;
                for (int ci = 0; ci < cnt; ++ci) mx = fmaxf(mx, logits[ci]);
                float se = 0.0f;
                for (int ci = 0; ci < cnt; ++ci) se += __expf(logits[ci] - mx);
                float lse = mx + __logf(se);
                int best = 0; float bp = -1e30f;
                for (int ci = 0; ci < cnt; ++ci) {
                    float pert = (logits[ci] - lse) + gumbel[(t * BATCH + rowg) * 5 + ci];
                    if (pert > bp) { bp = pert; best = ci; }
                }
                lp_acc += logits[best] - lse;
                chs[ncol] = best;
                out[t * BATCH + rowg] = (float)best;
            }
            __syncthreads();
            // x = emb[t][choice]  (all 32 lanes, 4 elems each per row)
            for (int r = 0; r < 16; ++r) {
                int cch = chs[r];
                float4 v = *(const float4*)(emb + (t * 5 + cch) * HDIM + lane * 4);
                bf16_t* xp = xbuf + r * HDIM + lane * 4;
                xp[0] = (bf16_t)v.x; xp[1] = (bf16_t)v.y;
                xp[2] = (bf16_t)v.z; xp[3] = (bf16_t)v.w;
            }
            __syncthreads();
        } else {
            // -------- continuous head: Gaussian sample + squash ------------
            int tc = t - NDSTEP;
            if (lane < 16) {
                int rowg = rowg0 + ncol;
                const float* w0 = ch_w + (tc * 2 + 0) * HDIM;
                const float* w1 = ch_w + (tc * 2 + 1) * HDIM;
                float a0 = 0.0f, a1 = 0.0f;
                for (int k = 0; k < HDIM; ++k) {
                    float o = (float)outrow[k];
                    a0 += o * w0[k]; a1 += o * w1[k];
                }
                float mean = ch_b[tc * 2 + 0] + a0;
                float lstd = ch_b[tc * 2 + 1] + a1;
                float ev  = eps[tc * BATCH + rowg];
                float std = __expf(lstd);
                float sample = mean + std * ev;
                lp_acc += -0.5f * ev * ev - lstd - 0.9189385332046727f;
                float lo = minv[tc], hi = maxv[tc];
                float p = sigm(sample) * (hi - lo) + lo;
                p = fminf(fmaxf(p, lo), hi);
                if (tc == 0 || tc == 1 || tc == 2 || tc == 5) p = rintf(p);
                out[(NDSTEP + tc) * BATCH + rowg] = p;
            }
            __syncthreads();
        }
    }
    if (lane < 16) out[13 * BATCH + rowg0 + ncol] = lp_acc;
}

// ---------------------------------------------------------------------------
extern "C" void kernel_launch(void* const* d_in, const int* in_sizes, int n_in,
                              void* d_out, int out_size, void* d_ws, size_t ws_size,
                              hipStream_t stream)
{
    const float* emb    = (const float*)d_in[0];
    const float* dh_w   = (const float*)d_in[1];
    const float* dh_b   = (const float*)d_in[2];
    const float* W_ih   = (const float*)d_in[3];
    const float* W_hh   = (const float*)d_in[4];
    const float* b_ih   = (const float*)d_in[5];
    const float* b_hh   = (const float*)d_in[6];
    const float* ch_w   = (const float*)d_in[7];
    const float* ch_b   = (const float*)d_in[8];
    const float* min_v  = (const float*)d_in[9];
    const float* max_v  = (const float*)d_in[10];
    const float* gumbel = (const float*)d_in[11];
    const float* eps    = (const float*)d_in[12];
    float*       outp   = (float*)d_out;

    bf16_t* wsw = (bf16_t*)d_ws;                              // 524288 B swizzled bf16 weights
    float*  wsb = (float*)((char*)d_ws + 512 * 1024);         // 4096 B combined bias

    ctrl_prep_kernel<<<1024, 256, 0, stream>>>(W_ih, W_hh, b_ih, b_hh, wsw, wsb);
    ctrl_main_kernel<<<BATCH / 16, 32, 0, stream>>>(emb, dh_w, dh_b, ch_w, ch_b,
                                                    min_v, max_v, gumbel, eps,
                                                    wsw, wsb, outp);
}
